// GNN_82008105550320
// MI455X (gfx1250) — matrix operand order repaired and checked by reading, not compiled, not run on hardware
//
#include <hip/hip_runtime.h>
#include <hip/hip_bf16.h>

// ---------------------------------------------------------------------------
// GNN pipeline for MI455X (gfx1250, wave32, WMMA).
//   h    = relu( segsum((x*ns)@W1)[dst] * nd + b1 )
//   z    =       segsum((h*ns)@W2)[dst] * nd + b2
//   out  = relu( [z[u],z[v]] @ Wc1 + bc1 ) @ Wc2 + bc2
// All GEMMs run on v_wmma_f32_16x16x32_f16 (f16 in, f32 accumulate).
// ---------------------------------------------------------------------------

typedef __attribute__((ext_vector_type(16))) _Float16 v16h;
typedef __attribute__((ext_vector_type(8)))  float    v8f;
typedef __attribute__((ext_vector_type(4)))  float    v4f;

static __device__ __forceinline__ v8f wmma_f16(v16h a, v16h b, v8f c) {
  // (neg_a, A, neg_b, B, c_mod, C, reuse_a, reuse_b)
  return __builtin_amdgcn_wmma_f32_16x16x32_f16(false, a, false, b, (short)0, c,
                                                false, false);
}

// ---------------------------------------------------------------------------
// Stage a [Ktot x 128] f32 weight matrix into LDS as f16, pre-swizzled into
// WMMA B-fragment order. Fragment (nt, kt) covers B[kt*32 .. kt*32+31][nt*16 ..
// nt*16+15]. Per ISA 16-bit B layout: lane<16 -> n=lane, K = k0 + h (h<8), k0+16+h-8 (h>=8);
// lane>=16 -> n=lane-16, same shifted by +8. Each lane's 16 halves are stored
// contiguously -> the WMMA-time read is one aligned 32B conflict-free ds load.
// ---------------------------------------------------------------------------
template <int KT32>  // number of 32-wide K steps (Ktot/32)
static __device__ __forceinline__ void fill_b_lds(_Float16* lds,
                                                  const float* __restrict__ W,
                                                  int tid, int nthreads) {
  const int total = 128 * KT32 * 32;  // = 128 * Ktot halves
  for (int idx = tid; idx < total; idx += nthreads) {
    const int h  = idx & 15;
    const int l  = (idx >> 4) & 31;
    const int fi = idx >> 9;             // fragment index = nt*KT32 + kt
    const int kt = fi % KT32;
    const int nt = fi / KT32;
    const int n  = nt * 16 + (l & 15);
    const int k  = kt * 32 + ((h & 8) ? 16 : 0) + ((l & 16) ? 8 : 0) + (h & 7);
    lds[idx] = (_Float16)W[k * 128 + n];  // W is [Ktot][128] row-major
  }
}

// Build an A fragment (16x32 f16, ISA layout) from one f32 row with a scale.
// lane<16: halves 0-7 = K k0..k0+7, halves 8-15 = K k0+16..k0+23.
// lane>=16: shift both runs by +8. 'off' = (lane>=16)*8.
static __device__ __forceinline__ v16h load_a_row(const float* __restrict__ row,
                                                  int k0, int off, float scale) {
  const v4f a0 = *(const v4f*)(row + k0 + off);
  const v4f a1 = *(const v4f*)(row + k0 + off + 4);
  const v4f a2 = *(const v4f*)(row + k0 + off + 16);
  const v4f a3 = *(const v4f*)(row + k0 + off + 20);
  v16h r;
#pragma unroll
  for (int i = 0; i < 4; ++i) {
    r[i]      = (_Float16)(a0[i] * scale);
    r[4 + i]  = (_Float16)(a1[i] * scale);
    r[8 + i]  = (_Float16)(a2[i] * scale);
    r[12 + i] = (_Float16)(a3[i] * scale);
  }
  return r;
}

// ---------------------------------------------------------------------------
// Small utility kernels
// ---------------------------------------------------------------------------
__global__ void k_zero_f32(float* __restrict__ p, int n) {
  const int t = blockIdx.x * blockDim.x + threadIdx.x;
  if (t < n) p[t] = 0.0f;
}

__global__ void k_degrees(const int* __restrict__ src, const int* __restrict__ dst,
                          unsigned* __restrict__ degOut, unsigned* __restrict__ degIn,
                          int nEdges) {
  const int t = blockIdx.x * blockDim.x + threadIdx.x;
  if (t >= nEdges) return;
  atomicAdd(&degOut[src[t]], 1u);
  atomicAdd(&degIn[dst[t]], 1u);
}

// Convert u32 degree counts -> float 1/sqrt(max(deg,1)) in place.
__global__ void k_norms(unsigned* __restrict__ degOut, unsigned* __restrict__ degIn,
                        int n) {
  const int t = blockIdx.x * blockDim.x + threadIdx.x;
  if (t >= n) return;
  const float a = fmaxf((float)degOut[t], 1.0f);
  const float b = fmaxf((float)degIn[t], 1.0f);
  ((float*)degOut)[t] = 1.0f / sqrtf(a);
  ((float*)degIn)[t]  = 1.0f / sqrtf(b);
}

// SpMM scatter: agg[dst[e]] += pre[src[e]]  (128 feats, 4 per thread).
__global__ void k_scatter(const float* __restrict__ pre, const int* __restrict__ src,
                          const int* __restrict__ dst, float* __restrict__ agg,
                          int nEdges) {
  const int t = blockIdx.x * blockDim.x + threadIdx.x;
  const int e = t >> 5;
  if (e >= nEdges) return;
  const int f = (t & 31) * 4;
  const float4 val = *(const float4*)(pre + (size_t)src[e] * 128 + f);
  float* dp = agg + (size_t)dst[e] * 128 + f;
  atomicAdd(dp + 0, val.x);
  atomicAdd(dp + 1, val.y);
  atomicAdd(dp + 2, val.z);
  atomicAdd(dp + 3, val.w);
}

// Epilogue: h = (agg * nd[row]) + b[col], optional relu.
template <bool RELU>
__global__ void k_post(const float* __restrict__ agg, const float* __restrict__ nd,
                       const float* __restrict__ b, float* __restrict__ h, int n) {
  const int t = blockIdx.x * blockDim.x + threadIdx.x;
  if (t >= n) return;
  const int i = t >> 7;
  const int f = t & 127;
  float val = agg[t] * nd[i] + b[f];
  if (RELU) val = fmaxf(val, 0.0f);
  h[t] = val;
}

// ---------------------------------------------------------------------------
// Node GEMM: C[m][n] = sum_k (A[m][k]*ns[m]) * W[k][n]; M=16*mtiles, N=K=128.
// One wave per 16-row M tile; 4 k-steps x 8 n-tiles = 32 WMMAs per tile.
// ---------------------------------------------------------------------------
__global__ void __launch_bounds__(256)
k_gemm_node(const float* __restrict__ A, const float* __restrict__ ns,
            const float* __restrict__ W, float* __restrict__ C, int mtiles) {
  extern __shared__ v16h lds_frag[];                 // 32 frags * 1024B = 32KB
  _Float16* lds = (_Float16*)lds_frag;
  fill_b_lds<4>(lds, W, threadIdx.x, blockDim.x);
  __syncthreads();

  const int lane = threadIdx.x & 31;
  const int wave = threadIdx.x >> 5;
  const int nw   = blockDim.x >> 5;
  const int lm   = lane & 15;
  const int hi   = lane >> 4;   // 0 | 1
  const int off  = hi * 8;      // A-fragment K offset

  for (int mt = blockIdx.x * nw + wave; mt < mtiles; mt += gridDim.x * nw) {
    const int m = mt * 16 + lm;
    const float* row = A + (size_t)m * 128;
    const float scale = ns[m];
    v8f acc[8] = {};
#pragma unroll
    for (int kt = 0; kt < 4; ++kt) {
      const v16h a = load_a_row(row, kt * 32, off, scale);
#pragma unroll
      for (int nt = 0; nt < 8; ++nt) {
        const v16h b = lds_frag[(nt * 4 + kt) * 32 + lane];
        acc[nt] = wmma_f16(a, b, acc[nt]);
      }
    }
    // D layout: vgpr r -> row r (lanes 0-15) / row r+8 (lanes 16-31), col = lm.
#pragma unroll
    for (int nt = 0; nt < 8; ++nt) {
      float* cp = C + (size_t)(mt * 16 + hi * 8) * 128 + nt * 16 + lm;
#pragma unroll
      for (int r = 0; r < 8; ++r) cp[(size_t)r * 128] = acc[nt][r];
    }
  }
}

// ---------------------------------------------------------------------------
// Fused decode: for a 16-pair tile,
//   A = [z[u[p]] , z[v[p]]]  (16 x 256 f32 -> f16)
//   H = relu(A @ Wc1 + bc1)  (16 x 128, stays in registers)
//   out[p] = H @ Wc2 + bc2   (shfl reduction)
// ---------------------------------------------------------------------------
__global__ void __launch_bounds__(256)
k_decode(const float* __restrict__ z, const int* __restrict__ u,
         const int* __restrict__ v, const float* __restrict__ Wc1,
         const float* __restrict__ bc1, const float* __restrict__ Wc2,
         const float* __restrict__ bc2p, float* __restrict__ out, int ptiles) {
  extern __shared__ v16h lds_frag[];                 // 64 frags * 1024B = 64KB
  _Float16* lds = (_Float16*)lds_frag;
  fill_b_lds<8>(lds, Wc1, threadIdx.x, blockDim.x);
  __syncthreads();

  const int lane = threadIdx.x & 31;
  const int wave = threadIdx.x >> 5;
  const int nw   = blockDim.x >> 5;
  const int lm   = lane & 15;
  const int hi   = lane >> 4;
  const int off  = hi * 8;

  float bc1v[8], wc2v[8];
#pragma unroll
  for (int nt = 0; nt < 8; ++nt) {
    bc1v[nt] = bc1[nt * 16 + lm];
    wc2v[nt] = Wc2[nt * 16 + lm];
  }
  const float bc2 = bc2p[0];

  for (int pt = blockIdx.x * nw + wave; pt < ptiles; pt += gridDim.x * nw) {
    const int p = pt * 16 + lm;
    const float* ru = z + (size_t)u[p] * 128;
    const float* rv = z + (size_t)v[p] * 128;
    v8f acc[8] = {};
#pragma unroll
    for (int kt = 0; kt < 8; ++kt) {
      const int c1 = kt * 32 + off;   // first 8-float run
      const int c2 = c1 + 16;        // second 8-float run (same 32-block)
      const float* s1 = (c1 < 128) ? (ru + c1) : (rv + (c1 - 128));
      const float* s2 = (c2 < 128) ? (ru + c2) : (rv + (c2 - 128));
      const v4f a0 = *(const v4f*)(s1);
      const v4f a1 = *(const v4f*)(s1 + 4);
      const v4f a2 = *(const v4f*)(s2);
      const v4f a3 = *(const v4f*)(s2 + 4);
      v16h a;
#pragma unroll
      for (int i = 0; i < 4; ++i) {
        a[i]      = (_Float16)a0[i];
        a[4 + i]  = (_Float16)a1[i];
        a[8 + i]  = (_Float16)a2[i];
        a[12 + i] = (_Float16)a3[i];
      }
#pragma unroll
      for (int nt = 0; nt < 8; ++nt) {
        const v16h b = lds_frag[(nt * 8 + kt) * 32 + lane];
        acc[nt] = wmma_f16(a, b, acc[nt]);
      }
    }
    // Second layer: out[row] = sum_n relu(H[row][n]) * Wc2[n] + bc2.
    // Lane holds rows (hi*8 .. hi*8+7) of column n = nt*16+lm per n-tile.
    float s[8];
#pragma unroll
    for (int r = 0; r < 8; ++r) s[r] = 0.0f;
#pragma unroll
    for (int nt = 0; nt < 8; ++nt) {
#pragma unroll
      for (int r = 0; r < 8; ++r) {
        float hval = acc[nt][r] + bc1v[nt];
        hval = hval > 0.0f ? hval : 0.0f;
        s[r] += hval * wc2v[nt];
      }
    }
    // Reduce across the 16 lanes of each half-wave (xor masks stay in half).
#pragma unroll
    for (int msk = 1; msk < 16; msk <<= 1) {
#pragma unroll
      for (int r = 0; r < 8; ++r) s[r] += __shfl_xor(s[r], msk, 32);
    }
#pragma unroll
    for (int r = 0; r < 8; ++r)
      if (lm == r) out[pt * 16 + hi * 8 + r] = s[r] + bc2;
  }
}

// ---------------------------------------------------------------------------
// Host-side launch sequence (all on `stream`, graph-capture safe)
// ---------------------------------------------------------------------------
extern "C" void kernel_launch(void* const* d_in, const int* in_sizes, int n_in,
                              void* d_out, int out_size, void* d_ws, size_t ws_size,
                              hipStream_t stream) {
  const float* x   = (const float*)d_in[0];
  const int*   src = (const int*)d_in[1];
  const int*   dst = (const int*)d_in[2];
  const int*   u   = (const int*)d_in[3];
  const int*   v   = (const int*)d_in[4];
  const float* W1  = (const float*)d_in[5];
  const float* b1  = (const float*)d_in[6];
  const float* W2  = (const float*)d_in[7];
  const float* b2  = (const float*)d_in[8];
  const float* Wc1 = (const float*)d_in[9];
  const float* bc1 = (const float*)d_in[10];
  const float* Wc2 = (const float*)d_in[11];
  const float* bc2 = (const float*)d_in[12];

  const int N      = in_sizes[0] / 128;
  const int E      = in_sizes[1];
  const int P      = in_sizes[3];
  const int NH     = N * 128;
  const int mtiles = N / 16;   // 50000/16 = 3125 exact
  const int ptiles = P / 16;   // 1e6/16  = 62500 exact

  float* ws  = (float*)d_ws;
  float* nsb = ws;                                     // N floats (deg_out -> ns)
  float* ndb = ws + N;                                 // N floats (deg_in  -> nd)
  const size_t oA = ((size_t)2 * N + 3) & ~(size_t)3;  // 16B-align
  float* bufA = ws + oA;                               // N*128 floats
  float* bufB = bufA + (size_t)NH;                     // N*128 floats

  const int T = 256;
  const int scatter_threads = E * 32;

  // Degrees + symmetric normalization.
  k_zero_f32<<<(2 * N + T - 1) / T, T, 0, stream>>>(nsb, 2 * N);
  k_degrees<<<(E + T - 1) / T, T, 0, stream>>>(src, dst, (unsigned*)nsb,
                                               (unsigned*)ndb, E);
  k_norms<<<(N + T - 1) / T, T, 0, stream>>>((unsigned*)nsb, (unsigned*)ndb, N);

  // Layer 1: hpre=(x*ns)@W1 -> scatter -> h=relu(agg*nd+b1)
  k_gemm_node<<<(mtiles + 7) / 8, T, 128 * 128 * 2, stream>>>(x, nsb, W1, bufA,
                                                              mtiles);
  k_zero_f32<<<(NH + T - 1) / T, T, 0, stream>>>(bufB, NH);
  k_scatter<<<(scatter_threads + T - 1) / T, T, 0, stream>>>(bufA, src, dst, bufB, E);
  k_post<true><<<(NH + T - 1) / T, T, 0, stream>>>(bufB, ndb, b1, bufA, NH);

  // Layer 2: zpre=(h*ns)@W2 -> scatter -> z=agg*nd+b2
  k_gemm_node<<<(mtiles + 7) / 8, T, 128 * 128 * 2, stream>>>(bufA, nsb, W2, bufB,
                                                              mtiles);
  k_zero_f32<<<(NH + T - 1) / T, T, 0, stream>>>(bufA, NH);
  k_scatter<<<(scatter_threads + T - 1) / T, T, 0, stream>>>(bufB, src, dst, bufA, E);
  k_post<false><<<(NH + T - 1) / T, T, 0, stream>>>(bufA, ndb, b2, bufB, NH);

  // Fused pair-MLP decode (dominant 65.5 GFLOP GEMM) -> d_out.
  int dblocks = (ptiles + 7) / 8;
  if (dblocks > 2048) dblocks = 2048;
  k_decode<<<dblocks, T, 256 * 128 * 2, stream>>>(bufB, u, v, Wc1, bc1, Wc2, bc2,
                                                  (float*)d_out, ptiles);

  (void)n_in; (void)out_size; (void)ws_size;
}
